// TextVAE_75754633167538
// MI455X (gfx1250) — compile-verified, matrix operand order checked by reading, not run on hardware
//
#include <hip/hip_runtime.h>
#include <math.h>

#define VSZ   50257
#define VPAD  50304        // pad vocab rows to multiple of 64 (4 N-tiles per wave)
#define EMB   256
#define HID   512
#define LAT   256
#define BATCH 32
#define SEQ   512
#define TDEC  64
#define KCAT  768          // EMB + HID (concatenated LSTM input)
#define NGATE 2048         // 4*HID
#define NGRP  (VPAD / 64)  // 786 wave-groups in logits GEMM

typedef __bf16 bf16_t;
typedef __attribute__((ext_vector_type(8)))  __bf16 v8bf;
typedef __attribute__((ext_vector_type(16))) __bf16 v16bf;
typedef __attribute__((ext_vector_type(8)))  float  v8f;

struct GBar { unsigned cnt; unsigned gen; };

// ---------------------------------------------------------------- helpers
__device__ __forceinline__ v8f zero8() {
  v8f z;
  for (int i = 0; i < 8; ++i) z[i] = 0.f;
  return z;
}

// Load one WMMA 16-lane-row fragment: 16 bf16 = two contiguous 16B chunks
// (K offsets g*8+{0..7} and 16+g*8+{0..7} handled by caller passing p=row+k0+g*8)
__device__ __forceinline__ v16bf load_frag(const bf16_t* p) {
  v8bf lo = *reinterpret_cast<const v8bf*>(p);
  v8bf hi = *reinterpret_cast<const v8bf*>(p + 16);
  return __builtin_shufflevector(lo, hi, 0,1,2,3,4,5,6,7,8,9,10,11,12,13,14,15);
}

__device__ __forceinline__ v8f wmma_bf16(v16bf a, v16bf b, v8f c) {
  return __builtin_amdgcn_wmma_f32_16x16x32_bf16(false, a, false, b, (short)0, c,
                                                 false, false);
}

__device__ __forceinline__ float sigmoidf_(float v) { return 1.f / (1.f + expf(-v)); }

// torch gate order i,f,g,o in 4 blocks of HID
__device__ __forceinline__ void lstm_act(const float* __restrict__ gates,
                                         const float* __restrict__ bc,
                                         float* __restrict__ c,
                                         bf16_t* __restrict__ h_out, int ldh, int i) {
  int b = i >> 9;          // /HID
  int j = i & (HID - 1);
  const float* gr = gates + (size_t)b * NGATE;
  float gi = gr[j]           + bc[j];
  float gf = gr[HID + j]     + bc[HID + j];
  float gg = gr[2*HID + j]   + bc[2*HID + j];
  float go = gr[3*HID + j]   + bc[3*HID + j];
  float cv = c[i];
  float cn = sigmoidf_(gf) * cv + sigmoidf_(gi) * tanhf(gg);
  float hn = sigmoidf_(go) * tanhf(cn);
  c[i] = cn;
  h_out[(size_t)b * ldh + j] = (bf16_t)hn;
}

// Device-wide generation barrier (16 co-resident blocks, agent scope)
__device__ __forceinline__ void grid_sync(GBar* bar, unsigned nblk) {
  __syncthreads();
  if (threadIdx.x == 0) {
    __threadfence();
    unsigned old = __hip_atomic_load(&bar->gen, __ATOMIC_ACQUIRE, __HIP_MEMORY_SCOPE_AGENT);
    unsigned a = __hip_atomic_fetch_add(&bar->cnt, 1u, __ATOMIC_ACQ_REL, __HIP_MEMORY_SCOPE_AGENT);
    if (a == nblk - 1u) {
      __hip_atomic_store(&bar->cnt, 0u, __ATOMIC_RELAXED, __HIP_MEMORY_SCOPE_AGENT);
      __hip_atomic_fetch_add(&bar->gen, 1u, __ATOMIC_ACQ_REL, __HIP_MEMORY_SCOPE_AGENT);
    } else {
      while (__hip_atomic_load(&bar->gen, __ATOMIC_ACQUIRE, __HIP_MEMORY_SCOPE_AGENT) == old)
        __builtin_amdgcn_s_sleep(1);
    }
    __threadfence();
  }
  __syncthreads();
}

// ---------------------------------------------------------------- prep kernels
__global__ void __launch_bounds__(256)
cat_w_kernel(const float* __restrict__ ih, const float* __restrict__ hh,
             bf16_t* __restrict__ dst) {       // dst [NGATE, KCAT] bf16
  int i = blockIdx.x * 256 + threadIdx.x;
  if (i >= NGATE * KCAT) return;
  int n = i / KCAT, k = i - n * KCAT;
  float v = (k < EMB) ? ih[(size_t)n * EMB + k] : hh[(size_t)n * HID + (k - EMB)];
  dst[i] = (bf16_t)v;
}

__global__ void __launch_bounds__(256)
ml_w_kernel(const float* __restrict__ mw, const float* __restrict__ lw,
            bf16_t* __restrict__ dst) {        // dst [2*LAT, HID] bf16
  int i = blockIdx.x * 256 + threadIdx.x;
  if (i >= 2 * LAT * HID) return;
  int n = i / HID, k = i - n * HID;
  float v = (n < LAT) ? mw[(size_t)n * HID + k] : lw[(size_t)(n - LAT) * HID + k];
  dst[i] = (bf16_t)v;
}

__global__ void __launch_bounds__(256)
outw_kernel(const float* __restrict__ w, bf16_t* __restrict__ dst) {  // [VPAD, HID]
  for (size_t i = (size_t)blockIdx.x * 256 + threadIdx.x; i < (size_t)VPAD * HID;
       i += (size_t)gridDim.x * 256) {
    size_t row = i / HID;
    dst[i] = (row < VSZ) ? (bf16_t)w[i] : (bf16_t)0.f;
  }
}

__global__ void __launch_bounds__(256)
gather_kernel(const int* __restrict__ x, const float* __restrict__ emb,
              bf16_t* __restrict__ xe) {       // xe [B, SEQ, EMB]
  size_t i = (size_t)blockIdx.x * 256 + threadIdx.x;
  if (i >= (size_t)BATCH * SEQ * EMB) return;
  int e  = (int)(i & (EMB - 1));
  int bt = (int)(i >> 8);                      // b*SEQ + t
  int tok = x[bt];
  xe[i] = (bf16_t)emb[(size_t)tok * EMB + e];
}

__global__ void __launch_bounds__(256)
misc_kernel(const float* ebih, const float* ebhh, const float* dbih, const float* dbhh,
            float* enc_bc, float* dec_bc, float* c_enc, bf16_t* hbf, GBar* bar) {
  int i = blockIdx.x * 256 + threadIdx.x;      // <<<64,256>>> = 16384 threads
  if (i < NGATE) { enc_bc[i] = ebih[i] + ebhh[i]; dec_bc[i] = dbih[i] + dbhh[i]; }
  if (i < BATCH * HID) { c_enc[i] = 0.f; hbf[i] = (bf16_t)0.f; }
  if (i == 0) { bar->cnt = 0u; bar->gen = 0u; }
}

// ---------------------------------------------------------------- persistent encoder
__global__ void __launch_bounds__(256)
vae_encoder_kernel(const bf16_t* __restrict__ xemb,   // [B, SEQ, EMB]
                   const bf16_t* __restrict__ Wcat,   // [NGATE, KCAT]
                   const float*  __restrict__ bc,     // [NGATE]
                   const bf16_t* __restrict__ Wml,    // [2*LAT, HID]
                   const float*  __restrict__ mean_b,
                   const float*  __restrict__ logvar_b,
                   const float*  __restrict__ eps,    // [B, LAT]
                   float*  __restrict__ gates,        // [B, NGATE]
                   float*  __restrict__ cbuf,         // [B, HID]
                   bf16_t* __restrict__ hbf,          // [B, HID]
                   float*  __restrict__ zml,          // [B, 2*LAT]
                   bf16_t* __restrict__ dec_in,       // [B, KCAT]
                   float*  __restrict__ c_dec,        // [B, HID]
                   GBar* bar) {
  const int lane = threadIdx.x & 31;
  const int ln   = lane & 15;
  const int g    = lane >> 4;
  const int wid  = (blockIdx.x << 3) + (threadIdx.x >> 5);  // 0..127
  const int tidg = blockIdx.x * 256 + threadIdx.x;          // 0..4095
  const unsigned nblk = gridDim.x;

  for (int t = 0; t < SEQ; ++t) {
    // ---- phase A: gates = [x_t | h] @ Wcat^T  (128 n-tiles, 2 m-accs each)
    {
      const int n0 = wid * 16;
      const bf16_t* brow = Wcat + (size_t)(n0 + ln) * KCAT;
      const bf16_t* a0x = xemb + ((size_t)ln        * SEQ + t) * EMB;
      const bf16_t* a1x = xemb + ((size_t)(ln + 16) * SEQ + t) * EMB;
      const bf16_t* a0h = hbf + (size_t)ln        * HID;
      const bf16_t* a1h = hbf + (size_t)(ln + 16) * HID;
      v8f acc0 = zero8(), acc1 = zero8();
      for (int k0 = 0; k0 < KCAT; k0 += 32) {
        v16bf bfr = load_frag(brow + k0 + g * 8);
        v16bf a0  = (k0 < EMB) ? load_frag(a0x + k0 + g * 8)
                               : load_frag(a0h + (k0 - EMB) + g * 8);
        v16bf a1  = (k0 < EMB) ? load_frag(a1x + k0 + g * 8)
                               : load_frag(a1h + (k0 - EMB) + g * 8);
        acc0 = wmma_bf16(a0, bfr, acc0);
        acc1 = wmma_bf16(a1, bfr, acc1);
      }
      for (int r = 0; r < 8; ++r) {
        gates[(size_t)(r + 8 * g)      * NGATE + n0 + ln] = acc0[r];
        gates[(size_t)(r + 8 * g + 16) * NGATE + n0 + ln] = acc1[r];
      }
    }
    grid_sync(bar, nblk);
    // ---- phase B: LSTM activation, write c (fp32) and h (bf16)
    for (int i = tidg; i < BATCH * HID; i += 4096)
      lstm_act(gates, bc, cbuf, hbf, HID, i);
    grid_sync(bar, nblk);
  }

  // ---- phase C: zml = hT @ [mean_W; logvar_W]^T   (64 tiles)
  if (wid < 64) {
    const int m0 = (wid & 1) * 16;
    const int n0 = (wid >> 1) * 16;
    const bf16_t* brow = Wml + (size_t)(n0 + ln) * HID;
    const bf16_t* arow = hbf + (size_t)(m0 + ln) * HID;
    v8f acc = zero8();
    for (int k0 = 0; k0 < HID; k0 += 32) {
      v16bf bfr = load_frag(brow + k0 + g * 8);
      v16bf afr = load_frag(arow + k0 + g * 8);
      acc = wmma_bf16(afr, bfr, acc);
    }
    for (int r = 0; r < 8; ++r)
      zml[(size_t)(m0 + r + 8 * g) * (2 * LAT) + n0 + ln] = acc[r];
  }
  grid_sync(bar, nblk);

  // ---- phase D: reparameterize z, init decoder input & state
  for (int i = tidg; i < BATCH * LAT; i += 4096) {
    int b = i >> 8, j = i & (LAT - 1);
    float zm = zml[(size_t)b * (2 * LAT) + j]       + mean_b[j];
    float lv = zml[(size_t)b * (2 * LAT) + LAT + j] + logvar_b[j];
    float z  = zm + eps[(size_t)b * LAT + j] * expf(0.5f * lv);
    dec_in[(size_t)b * KCAT + j]             = (bf16_t)z;     // z_i part
    dec_in[(size_t)b * KCAT + EMB + j]       = (bf16_t)0.f;   // h0 part (cols 256..511)
    dec_in[(size_t)b * KCAT + EMB + LAT + j] = (bf16_t)0.f;   // h0 part (cols 512..767)
    c_dec[(size_t)b * HID + j]       = 0.f;
    c_dec[(size_t)b * HID + LAT + j] = 0.f;
  }
}

// ---------------------------------------------------------------- decoder kernels
__global__ void __launch_bounds__(256)
dec_gates_kernel(const bf16_t* __restrict__ dec_in, const bf16_t* __restrict__ Wcat,
                 float* __restrict__ gates) {
  const int lane = threadIdx.x & 31, ln = lane & 15, g = lane >> 4;
  const int wid = (blockIdx.x << 3) + (threadIdx.x >> 5);
  const int n0 = wid * 16;
  const bf16_t* brow = Wcat + (size_t)(n0 + ln) * KCAT;
  const bf16_t* a0 = dec_in + (size_t)ln        * KCAT;
  const bf16_t* a1 = dec_in + (size_t)(ln + 16) * KCAT;
  v8f acc0 = zero8(), acc1 = zero8();
  for (int k0 = 0; k0 < KCAT; k0 += 32) {
    v16bf bfr = load_frag(brow + k0 + g * 8);
    acc0 = wmma_bf16(load_frag(a0 + k0 + g * 8), bfr, acc0);
    acc1 = wmma_bf16(load_frag(a1 + k0 + g * 8), bfr, acc1);
  }
  for (int r = 0; r < 8; ++r) {
    gates[(size_t)(r + 8 * g)      * NGATE + n0 + ln] = acc0[r];
    gates[(size_t)(r + 8 * g + 16) * NGATE + n0 + ln] = acc1[r];
  }
}

__global__ void __launch_bounds__(256)
dec_act_kernel(const float* __restrict__ gates, const float* __restrict__ bc,
               float* __restrict__ c, bf16_t* __restrict__ dec_in) {
  int i = blockIdx.x * 256 + threadIdx.x;      // <<<64,256>>> = B*HID
  lstm_act(gates, bc, c, dec_in + EMB, KCAT, i);  // write h into dec_in h-slot
}

// logits: each wave computes a 32x64 output block (2 M-accs x 4 N-tiles).
// Per K-step: 2 A-frag loads + 4 B-frag loads = 12 b128 for 8 WMMAs.
__global__ void __launch_bounds__(256)
dec_logits_kernel(const bf16_t* __restrict__ dec_in, const bf16_t* __restrict__ Wo,
                  const float* __restrict__ out_b, float* __restrict__ out, int t) {
  const int lane = threadIdx.x & 31, ln = lane & 15, g = lane >> 4;
  const int wid = (blockIdx.x << 3) + (threadIdx.x >> 5);
  if (wid >= NGRP) return;
  const int nbase = wid * 64;
  const bf16_t* brow0 = Wo + (size_t)(nbase + 0  + ln) * HID;
  const bf16_t* brow1 = Wo + (size_t)(nbase + 16 + ln) * HID;
  const bf16_t* brow2 = Wo + (size_t)(nbase + 32 + ln) * HID;
  const bf16_t* brow3 = Wo + (size_t)(nbase + 48 + ln) * HID;
  const bf16_t* a0 = dec_in + (size_t)ln        * KCAT + EMB;
  const bf16_t* a1 = dec_in + (size_t)(ln + 16) * KCAT + EMB;
  v8f acc00 = zero8(), acc01 = zero8();
  v8f acc10 = zero8(), acc11 = zero8();
  v8f acc20 = zero8(), acc21 = zero8();
  v8f acc30 = zero8(), acc31 = zero8();
  for (int k0 = 0; k0 < HID; k0 += 32) {
    v16bf af0 = load_frag(a0 + k0 + g * 8);
    v16bf af1 = load_frag(a1 + k0 + g * 8);
    v16bf b0 = load_frag(brow0 + k0 + g * 8);
    acc00 = wmma_bf16(af0, b0, acc00);
    acc01 = wmma_bf16(af1, b0, acc01);
    v16bf b1 = load_frag(brow1 + k0 + g * 8);
    acc10 = wmma_bf16(af0, b1, acc10);
    acc11 = wmma_bf16(af1, b1, acc11);
    v16bf b2 = load_frag(brow2 + k0 + g * 8);
    acc20 = wmma_bf16(af0, b2, acc20);
    acc21 = wmma_bf16(af1, b2, acc21);
    v16bf b3 = load_frag(brow3 + k0 + g * 8);
    acc30 = wmma_bf16(af0, b3, acc30);
    acc31 = wmma_bf16(af1, b3, acc31);
  }
  v8f* accs[4][2] = {{&acc00, &acc01}, {&acc10, &acc11},
                     {&acc20, &acc21}, {&acc30, &acc31}};
  for (int nt = 0; nt < 4; ++nt) {
    int n = nbase + nt * 16 + ln;
    if (n < VSZ) {
      float bias = out_b[n];
      for (int r = 0; r < 8; ++r) {
        int m = r + 8 * g;
        out[((size_t)m        * TDEC + t) * VSZ + n] = (*accs[nt][0])[r] + bias;
        out[((size_t)(m + 16) * TDEC + t) * VSZ + n] = (*accs[nt][1])[r] + bias;
      }
    }
  }
}

__global__ void __launch_bounds__(256)
dec_argmax_kernel(const float* __restrict__ out, const float* __restrict__ embedding,
                  bf16_t* __restrict__ dec_in, int t) {
  const int b = blockIdx.x;
  const float* row = out + ((size_t)b * TDEC + t) * VSZ;
  float best = -INFINITY; int bidx = 0x7fffffff;
  for (int v = threadIdx.x; v < VSZ; v += 256) {
    float xv = row[v];
    if (xv > best) { best = xv; bidx = v; }    // strict > = first occurrence per slice
  }
  __shared__ float sv[256];
  __shared__ int   si[256];
  sv[threadIdx.x] = best; si[threadIdx.x] = bidx;
  __syncthreads();
  for (int s = 128; s > 0; s >>= 1) {
    if (threadIdx.x < s) {
      float ov = sv[threadIdx.x + s]; int oi = si[threadIdx.x + s];
      if (ov > sv[threadIdx.x] || (ov == sv[threadIdx.x] && oi < si[threadIdx.x])) {
        sv[threadIdx.x] = ov; si[threadIdx.x] = oi;
      }
    }
    __syncthreads();
  }
  int idx = si[0];                              // next-step input: embedding[pred]
  dec_in[(size_t)b * KCAT + threadIdx.x] = (bf16_t)embedding[(size_t)idx * EMB + threadIdx.x];
}

// ---------------------------------------------------------------- host launcher
extern "C" void kernel_launch(void* const* d_in, const int* in_sizes, int n_in,
                              void* d_out, int out_size, void* d_ws, size_t ws_size,
                              hipStream_t stream) {
  (void)in_sizes; (void)n_in; (void)out_size; (void)ws_size;
  const int*   x         = (const int*)  d_in[0];
  const float* eps       = (const float*)d_in[1];
  const float* embedding = (const float*)d_in[2];
  const float* enc_Wih   = (const float*)d_in[3];
  const float* enc_Whh   = (const float*)d_in[4];
  const float* enc_bih   = (const float*)d_in[5];
  const float* enc_bhh   = (const float*)d_in[6];
  const float* mean_W    = (const float*)d_in[7];
  const float* mean_b    = (const float*)d_in[8];
  const float* logvar_W  = (const float*)d_in[9];
  const float* logvar_b  = (const float*)d_in[10];
  const float* dec_Wih   = (const float*)d_in[11];
  const float* dec_Whh   = (const float*)d_in[12];
  const float* dec_bih   = (const float*)d_in[13];
  const float* dec_bhh   = (const float*)d_in[14];
  const float* out_W     = (const float*)d_in[15];
  const float* out_b     = (const float*)d_in[16];
  float* out = (float*)d_out;

  char* p = (char*)d_ws;
  auto carve = [&](size_t bytes) -> void* {
    void* r = (void*)p; p += (bytes + 255) & ~(size_t)255; return r;
  };
  bf16_t* WencCat = (bf16_t*)carve((size_t)NGATE * KCAT * 2);
  bf16_t* WdecCat = (bf16_t*)carve((size_t)NGATE * KCAT * 2);
  bf16_t* Wml     = (bf16_t*)carve((size_t)2 * LAT * HID * 2);
  bf16_t* outWbf  = (bf16_t*)carve((size_t)VPAD * HID * 2);
  bf16_t* xemb    = (bf16_t*)carve((size_t)BATCH * SEQ * EMB * 2);
  float*  enc_bc  = (float*) carve((size_t)NGATE * 4);
  float*  dec_bc  = (float*) carve((size_t)NGATE * 4);
  float*  gates   = (float*) carve((size_t)BATCH * NGATE * 4);
  float*  c_enc   = (float*) carve((size_t)BATCH * HID * 4);
  float*  c_dec   = (float*) carve((size_t)BATCH * HID * 4);
  bf16_t* hbf     = (bf16_t*)carve((size_t)BATCH * HID * 2);
  float*  zml     = (float*) carve((size_t)BATCH * 2 * LAT * 4);
  bf16_t* dec_in  = (bf16_t*)carve((size_t)BATCH * KCAT * 2);
  GBar*   bar     = (GBar*)  carve(256);

  // ---- prep (re-runs every call; fully re-initializes workspace) ----
  cat_w_kernel<<<(NGATE * KCAT + 255) / 256, 256, 0, stream>>>(enc_Wih, enc_Whh, WencCat);
  cat_w_kernel<<<(NGATE * KCAT + 255) / 256, 256, 0, stream>>>(dec_Wih, dec_Whh, WdecCat);
  ml_w_kernel<<<(2 * LAT * HID + 255) / 256, 256, 0, stream>>>(mean_W, logvar_W, Wml);
  outw_kernel<<<4096, 256, 0, stream>>>(out_W, outWbf);
  gather_kernel<<<(BATCH * SEQ * EMB + 255) / 256, 256, 0, stream>>>(x, embedding, xemb);
  misc_kernel<<<64, 256, 0, stream>>>(enc_bih, enc_bhh, dec_bih, dec_bhh,
                                      enc_bc, dec_bc, c_enc, hbf, bar);

  // ---- persistent encoder (512 recurrent steps + z GEMM + reparam) ----
  vae_encoder_kernel<<<16, 256, 0, stream>>>(xemb, WencCat, enc_bc, Wml, mean_b,
                                             logvar_b, eps, gates, c_enc, hbf, zml,
                                             dec_in, c_dec, bar);

  // ---- autoregressive decoder: 64 steps x 4 kernels ----
  for (int t = 0; t < TDEC; ++t) {
    dec_gates_kernel<<<16, 256, 0, stream>>>(dec_in, WdecCat, gates);
    dec_act_kernel<<<64, 256, 0, stream>>>(gates, dec_bc, c_dec, dec_in);
    dec_logits_kernel<<<(NGRP + 7) / 8, 256, 0, stream>>>(dec_in, outWbf, out_b, out, t);
    if (t + 1 < TDEC)
      dec_argmax_kernel<<<BATCH, 256, 0, stream>>>(out, embedding, dec_in, t);
  }
}
